// DecoderLayer_23656679866666
// MI455X (gfx1250) — matrix operand order repaired and checked by reading, not compiled, not run on hardware
//
#include <hip/hip_runtime.h>

#define B_   4
#define S_   1024
#define H_   2048
#define NH_  16
#define KVH_ 4
#define HD_  128
#define I_   8192
#define TOK_ (B_*S_)   // 4096 tokens

#define LDA_ 40        // padded LDS row stride (elements) -> 80B, conflict-free
                       // (matches TDM pad: 16 DWORDs data + 4 DWORDs pad per row)

#if defined(__has_builtin)
#  if __has_builtin(__builtin_amdgcn_tensor_load_to_lds)
#    define HAVE_TDM 1
#  endif
#endif
#ifndef HAVE_TDM
#  define HAVE_TDM 0
#endif

typedef __attribute__((ext_vector_type(16))) __bf16       v16bf;
typedef __attribute__((ext_vector_type(8)))  float        v8f;
typedef __attribute__((ext_vector_type(4)))  unsigned int v4u;
typedef __attribute__((ext_vector_type(8)))  int          i32x8;
typedef __attribute__((ext_vector_type(4)))  int          i32x4;
typedef unsigned short u16;

union FragU { v16bf v; v4u q[2]; u16 s[16]; };

static __device__ __forceinline__ u16 f32_to_bf16_bits(float x) {
  unsigned u = __float_as_uint(x);
  u += 0x7FFFu + ((u >> 16) & 1u);          // round-to-nearest-even
  return (u16)(u >> 16);
}

static __device__ __forceinline__ v8f v8f_zero() {
  v8f r;
  #pragma unroll
  for (int i = 0; i < 8; ++i) r[i] = 0.0f;
  return r;
}

// A-fragment (16 rows x 32 K, bf16), row-major source with leading dim ld.
// lane: row = lane&15 ; K run: lanes0-15 -> K 0..15, lanes16-31 -> K 16..31 (contiguous)
static __device__ __forceinline__ v16bf load_a_frag(const u16* base, int ld) {
  int lane = threadIdx.x & 31;
  const u16* p = base + (size_t)(lane & 15) * ld + ((lane >> 4) << 4);
  FragU f;
  f.q[0] = *(const v4u*)(p);
  f.q[1] = *(const v4u*)(p + 8);
  return f.v;
}

// B-fragment (32 K x 16 cols, bf16) from row-major W[n][k] storage:
// lane: col n = lane&15 ; K run: lanes0-15 -> K 0..15, lanes16-31 -> K 16..31 (contiguous)
static __device__ __forceinline__ v16bf load_b_frag(const u16* base, int ld) {
  int lane = threadIdx.x & 31;
  const u16* p = base + (size_t)(lane & 15) * ld + ((lane >> 4) << 4);
  FragU f;
  f.q[0] = *(const v4u*)(p);
  f.q[1] = *(const v4u*)(p + 8);
  return f.v;
}

// B-fragment from K-major storage (V[k][n], leading dim ld): strided gather.
static __device__ __forceinline__ v16bf load_b_frag_strided(const u16* base, int ld) {
  int lane = threadIdx.x & 31;
  int n  = lane & 15;
  int kb = (lane >> 4) << 4;
  FragU f;
  #pragma unroll
  for (int e = 0; e < 16; ++e) f.s[e] = base[(size_t)(kb + e) * ld + n];
  return f.v;
}

static __device__ __forceinline__ v8f wmma_bf16(v16bf a, v16bf b, v8f c) {
  return __builtin_amdgcn_wmma_f32_16x16x32_bf16(false, a, false, b, (short)0, c, false, false);
}

#if HAVE_TDM
// Issue a TDM load of a 2-D tile (32 K-elems wide x 128 rows, bf16) from a
// row-major tensor (leading dim K elems) into LDS at lds_addr, padding each
// 64B row with 16B so the LDS layout has stride LDA_=40 elements.
// D# layout per CDNA5 ISA ch.8 (group0 128b, group1 256b, groups 2/3 zero).
// This toolchain's builtin takes 6 args (g0, g1, g2, g3, g4, cpol).
static __device__ __forceinline__ void tdm_load_tile(const u16* gaddr,
                                                     unsigned lds_addr, int K) {
  unsigned long long ga = (unsigned long long)(size_t)gaddr;
  v4u g0;
  g0[0] = 1u;                                          // count=1, user mode
  g0[1] = lds_addr;                                    // LDS byte address
  g0[2] = (unsigned)(ga & 0xFFFFFFFFu);                // global_addr[31:0]
  g0[3] = (unsigned)((ga >> 32) & 0x01FFFFFFu)         // global_addr[56:32]
        | (2u << 30);                                  // type = 2 ("image")
  i32x8 g1;
  g1[0] = (1 << 16)      // data_size = 1 -> 2 bytes
        | (1 << 20)      // pad_enable
        | (3 << 22)      // pad_interval = 16 DWORDs (64B = one tile row)
        | (3 << 25);     // pad_amount   = 4 DWORDs (16B)
  g1[1] = (K & 0xFFFF) << 16;                          // tensor_dim0[15:0]
  g1[2] = ((unsigned)K >> 16) | (128u << 16);          // dim0 hi | tensor_dim1=128
  g1[3] = (32 << 16);                                  // tile_dim0 = 32
  g1[4] = 128;                                         // tile_dim1 = 128, tile_dim2=0
  g1[5] = K;                                           // tensor_dim0_stride lo
  g1[6] = 0;                                           // stride hi | dim1_stride lo
  g1[7] = 0;
  i32x4 z4;
  z4[0] = 0; z4[1] = 0; z4[2] = 0; z4[3] = 0;
  i32x8 z8;
  #pragma unroll
  for (int i = 0; i < 8; ++i) z8[i] = 0;
  __builtin_amdgcn_tensor_load_to_lds(g0, g1, z4, z4, z8, 0);
}
#endif

// ---------------------------------------------------------------- elementwise
__global__ void cvt_f32_to_bf16_kernel(const float* __restrict__ in,
                                       u16* __restrict__ out, long n) {
  long i = (long)blockIdx.x * blockDim.x + threadIdx.x;
  if (i < n) out[i] = f32_to_bf16_bits(in[i]);
}

__global__ __launch_bounds__(256)
void rmsnorm_bf16_kernel(const float* __restrict__ x, const float* __restrict__ w,
                         u16* __restrict__ out) {
  int row = blockIdx.x;
  const float* xp = x + (size_t)row * H_;
  float ss = 0.f;
  for (int i = threadIdx.x; i < H_; i += 256) { float v = xp[i]; ss += v * v; }
  __shared__ float red[256];
  red[threadIdx.x] = ss;
  __syncthreads();
  for (int s = 128; s > 0; s >>= 1) {
    if ((int)threadIdx.x < s) red[threadIdx.x] += red[threadIdx.x + s];
    __syncthreads();
  }
  float rinv = rsqrtf(red[0] / (float)H_ + 1e-6f);
  u16* op = out + (size_t)row * H_;
  for (int i = threadIdx.x; i < H_; i += 256)
    op[i] = f32_to_bf16_bits(xp[i] * rinv * w[i]);
}

// RoPE + head transpose: x (B,S,nh,HD) f32 -> outT (B,nh,S,HD) bf16
__global__ void rope_transpose_kernel(const float* __restrict__ x,
                                      const float* __restrict__ sin_t,
                                      const float* __restrict__ cos_t,
                                      u16* __restrict__ outT, int nheads) {
  long idx = (long)blockIdx.x * blockDim.x + threadIdx.x;
  long total = (long)B_ * S_ * nheads * (HD_ / 2);
  if (idx >= total) return;
  int d = (int)(idx & (HD_ / 2 - 1));
  long t = idx >> 6;                 // HD/2 = 64
  int h = (int)(t % nheads); t /= nheads;
  int s = (int)(t % S_);     int b = (int)(t / S_);
  const float* xp = x + (((size_t)(b * S_ + s) * nheads + h) * HD_);
  float x1 = xp[d], x2 = xp[d + HD_ / 2];
  float c  = cos_t[(size_t)s * (HD_ / 2) + d];
  float sn = sin_t[(size_t)s * (HD_ / 2) + d];
  u16* op = outT + (((size_t)(b * nheads + h) * S_ + s) * HD_);
  op[d]            = f32_to_bf16_bits(x1 * c - x2 * sn);
  op[d + HD_ / 2]  = f32_to_bf16_bits(x2 * c + x1 * sn);
}

// transpose only (for V): x (B,S,KVH,HD) f32 -> outT (B,KVH,S,HD) bf16
__global__ void transpose_bf16_kernel(const float* __restrict__ x,
                                      u16* __restrict__ outT) {
  long idx = (long)blockIdx.x * blockDim.x + threadIdx.x;
  long total = (long)B_ * S_ * KVH_ * HD_;
  if (idx >= total) return;
  int d = (int)(idx & (HD_ - 1));
  long t = idx >> 7;                 // HD = 128
  int h = (int)(t % KVH_); t /= KVH_;
  int s = (int)(t % S_);   int b = (int)(t / S_);
  outT[(((size_t)(b * KVH_ + h) * S_ + s) * HD_) + d] =
      f32_to_bf16_bits(x[(((size_t)(b * S_ + s) * KVH_ + h) * HD_) + d]);
}

__global__ void silu_mul_kernel(const float* __restrict__ gu, u16* __restrict__ act) {
  long idx = (long)blockIdx.x * blockDim.x + threadIdx.x;
  long total = (long)TOK_ * I_;
  if (idx >= total) return;
  long t = idx / I_, i = idx % I_;
  float g = gu[t * (2L * I_) + i];
  float u = gu[t * (2L * I_) + I_ + i];
  float s = g / (1.0f + __expf(-g));
  act[idx] = f32_to_bf16_bits(s * u);
}

// ---------------------------------------------------------------- GEMM
// C[M,N](f32, +optional residual) = A[M,K](bf16) * W[N,K]^T(bf16)
// Block: 256 thr = 8 waves (2x4); wave tile 64x32 (4x2 WMMA tiles); block tile 128x128.
// Double-buffered LDS staging. When the TDM builtin is available, wave 0 drives
// tensor_load_to_lds DMAs (TENSORcnt) for the next K tile while all waves run
// WMMA on the current one; otherwise a cooperative vector-load path is used.
__global__ __launch_bounds__(256)
void gemm_bf16_kernel(const u16* __restrict__ A, const u16* __restrict__ Bw,
                      float* __restrict__ C, const float* __restrict__ residual,
                      int M, int N, int K) {
  __shared__ __align__(16) u16 As[2][128 * LDA_];
  __shared__ __align__(16) u16 Bs[2][128 * LDA_];

  int tid  = threadIdx.x;
  int wave = tid >> 5;
  int lane = tid & 31;
  int wm = wave >> 2;               // 0..1
  int wn = wave & 3;                // 0..3
  int m0 = blockIdx.y * 128;
  int n0 = blockIdx.x * 128;

  v8f acc[4][2];
  #pragma unroll
  for (int i = 0; i < 4; ++i)
    #pragma unroll
    for (int j = 0; j < 2; ++j) acc[i][j] = v8f_zero();

  auto compute = [&](int buf) {
    v16bf af[4], bf[2];
    #pragma unroll
    for (int i = 0; i < 4; ++i)
      af[i] = load_a_frag(&As[buf][(wm * 64 + 16 * i) * LDA_], LDA_);
    #pragma unroll
    for (int j = 0; j < 2; ++j)
      bf[j] = load_b_frag(&Bs[buf][(wn * 32 + 16 * j) * LDA_], LDA_);
    #pragma unroll
    for (int i = 0; i < 4; ++i)
      #pragma unroll
      for (int j = 0; j < 2; ++j)
        acc[i][j] = wmma_bf16(af[i], bf[j], acc[i][j]);
  };

#if HAVE_TDM
  const u16* Abase = A  + (size_t)m0 * K;
  const u16* Bbase = Bw + (size_t)n0 * K;
  unsigned ldsA0 = (unsigned)(size_t)&As[0][0];
  unsigned ldsA1 = (unsigned)(size_t)&As[1][0];
  unsigned ldsB0 = (unsigned)(size_t)&Bs[0][0];
  unsigned ldsB1 = (unsigned)(size_t)&Bs[1][0];

  if (wave == 0) {
    tdm_load_tile(Abase, ldsA0, K);
    tdm_load_tile(Bbase, ldsB0, K);
    __builtin_amdgcn_s_wait_tensorcnt(0);
  }
  __syncthreads();
  int buf = 0;
  for (int k0 = 32; k0 < K; k0 += 32) {
    if (wave == 0) {                               // DMA next tile (TENSORcnt)
      tdm_load_tile(Abase + k0, buf ? ldsA0 : ldsA1, K);
      tdm_load_tile(Bbase + k0, buf ? ldsB0 : ldsB1, K);
    }
    compute(buf);                                  // WMMA from LDS (current)
    if (wave == 0) __builtin_amdgcn_s_wait_tensorcnt(0);
    __syncthreads();
    buf ^= 1;
  }
  compute(buf);
#else
  v4u a_reg[2], b_reg[2];
  auto load_tile = [&](int k0) {
    #pragma unroll
    for (int r = 0; r < 2; ++r) {
      int linear = tid + r * 256;
      int row = linear >> 2;        // 0..127
      int ch  = (linear & 3) << 3;  // 0,8,16,24
      a_reg[r] = *(const v4u*)(A  + (size_t)(m0 + row) * K + k0 + ch);
      b_reg[r] = *(const v4u*)(Bw + (size_t)(n0 + row) * K + k0 + ch);
    }
  };
  auto store_tile = [&](int buf) {
    #pragma unroll
    for (int r = 0; r < 2; ++r) {
      int linear = tid + r * 256;
      int row = linear >> 2;
      int ch  = (linear & 3) << 3;
      *(v4u*)(&As[buf][row * LDA_ + ch]) = a_reg[r];
      *(v4u*)(&Bs[buf][row * LDA_ + ch]) = b_reg[r];
    }
  };
  load_tile(0);
  store_tile(0);
  __syncthreads();
  int buf = 0;
  for (int k0 = 32; k0 < K; k0 += 32) {
    load_tile(k0);
    if (k0 + 32 < K) {
      int row = tid >> 1;
      __builtin_prefetch(A  + (size_t)(m0 + row) * K + k0 + 32, 0, 1);
      __builtin_prefetch(Bw + (size_t)(n0 + row) * K + k0 + 32, 0, 1);
    }
    compute(buf);
    store_tile(buf ^ 1);
    __syncthreads();
    buf ^= 1;
  }
  compute(buf);
#endif

  int rb = (lane >> 4) << 3;
  #pragma unroll
  for (int i = 0; i < 4; ++i)
    #pragma unroll
    for (int j = 0; j < 2; ++j) {
      size_t base = (size_t)(m0 + wm * 64 + 16 * i + rb) * N
                  + (n0 + wn * 32 + 16 * j + (lane & 15));
      if (residual) {
        #pragma unroll
        for (int v = 0; v < 8; ++v)
          C[base + (size_t)v * N] = acc[i][j][v] + residual[base + (size_t)v * N];
      } else {
        #pragma unroll
        for (int v = 0; v < 8; ++v)
          C[base + (size_t)v * N] = acc[i][j][v];
      }
    }
}

// ---------------------------------------------------------------- flash attention
// One wave per (b, h, 16-query tile); online softmax; 32 keys per step.
__global__ __launch_bounds__(32)
void flash_attn_kernel(const u16* __restrict__ qT, const u16* __restrict__ kT,
                       const u16* __restrict__ vT, u16* __restrict__ attn_out) {
  __shared__ __align__(16) u16 p_lds[16 * 32];
  int q0 = blockIdx.x * 16;
  int h  = blockIdx.y;
  int b  = blockIdx.z;
  int hk = h >> 2;                   // NH/KVH = 4 (GQA)
  int lane = threadIdx.x;
  int rb   = (lane >> 4) << 3;

  const u16* qbase = qT + (((size_t)(b * NH_  + h ) * S_ + q0) * HD_);
  const u16* kbase = kT + (((size_t)(b * KVH_ + hk) * S_      ) * HD_);
  const u16* vbase = vT + (((size_t)(b * KVH_ + hk) * S_      ) * HD_);

  v16bf qf[4];
  #pragma unroll
  for (int c = 0; c < 4; ++c) qf[c] = load_a_frag(qbase + c * 32, HD_);

  v8f ov[8];
  float m[8], l[8];
  #pragma unroll
  for (int j = 0; j < 8; ++j) ov[j] = v8f_zero();
  #pragma unroll
  for (int v = 0; v < 8; ++v) { m[v] = -1e30f; l[v] = 0.0f; }

  const float scale = 0.08838834764831845f;   // 1/sqrt(128)

  for (int n0 = 0; n0 < q0 + 16; n0 += 32) {
    if (n0 + 32 < q0 + 16) {                  // per-lane row prefetch of next tile
      __builtin_prefetch(kbase + (size_t)(n0 + 32 + lane) * HD_, 0, 1);
      __builtin_prefetch(vbase + (size_t)(n0 + 32 + lane) * HD_, 0, 1);
    }
    v8f s0 = v8f_zero(), s1 = v8f_zero();
    #pragma unroll
    for (int c = 0; c < 4; ++c) {
      v16bf kf0 = load_b_frag(kbase + (size_t)(n0     ) * HD_ + c * 32, HD_);
      v16bf kf1 = load_b_frag(kbase + (size_t)(n0 + 16) * HD_ + c * 32, HD_);
      s0 = wmma_bf16(qf[c], kf0, s0);
      s1 = wmma_bf16(qf[c], kf1, s1);
    }

    int col0 = n0 + (lane & 15);
    int col1 = col0 + 16;
    float alpha[8];
    #pragma unroll
    for (int v = 0; v < 8; ++v) {
      int row = q0 + rb + v;
      float a  = s0[v] * scale; if (col0 > row) a  = -1e30f;
      float bq = s1[v] * scale; if (col1 > row) bq = -1e30f;
      float t = fmaxf(a, bq);
      t = fmaxf(t, __shfl_xor(t, 1, 32));
      t = fmaxf(t, __shfl_xor(t, 2, 32));
      t = fmaxf(t, __shfl_xor(t, 4, 32));
      t = fmaxf(t, __shfl_xor(t, 8, 32));     // row max within 16-lane half
      float nm = fmaxf(m[v], t);
      alpha[v] = __expf(m[v] - nm);
      m[v] = nm;
      float p0 = __expf(a  - nm);
      float p1 = __expf(bq - nm);
      p_lds[(rb + v) * 32 + (lane & 15)]      = f32_to_bf16_bits(p0);
      p_lds[(rb + v) * 32 + (lane & 15) + 16] = f32_to_bf16_bits(p1);
      float rs = p0 + p1;
      rs += __shfl_xor(rs, 1, 32);
      rs += __shfl_xor(rs, 2, 32);
      rs += __shfl_xor(rs, 4, 32);
      rs += __shfl_xor(rs, 8, 32);
      l[v] = l[v] * alpha[v] + rs;
    }
    #pragma unroll
    for (int j = 0; j < 8; ++j)
      #pragma unroll
      for (int v = 0; v < 8; ++v) ov[j][v] *= alpha[v];

    __syncthreads();                           // LDS write -> A-frag reload
    v16bf pf = load_a_frag(p_lds, 32);
    #pragma unroll
    for (int j = 0; j < 8; ++j) {
      v16bf vf = load_b_frag_strided(vbase + (size_t)n0 * HD_ + j * 16, HD_);
      ov[j] = wmma_bf16(pf, vf, ov[j]);
    }
    __syncthreads();
  }

  #pragma unroll
  for (int j = 0; j < 8; ++j)
    #pragma unroll
    for (int v = 0; v < 8; ++v) {
      int row = q0 + rb + v;
      float o = ov[j][v] / l[v];
      attn_out[(size_t)(b * S_ + row) * (NH_ * HD_) + h * HD_ + j * 16 + (lane & 15)] =
          f32_to_bf16_bits(o);
    }
}

// ---------------------------------------------------------------- host driver
extern "C" void kernel_launch(void* const* d_in, const int* in_sizes, int n_in,
                              void* d_out, int out_size, void* d_ws, size_t ws_size,
                              hipStream_t stream) {
  (void)in_sizes; (void)n_in; (void)out_size; (void)ws_size;
  const float* hidden = (const float*)d_in[0];
  const float* sin_t  = (const float*)d_in[1];
  const float* cos_t  = (const float*)d_in[2];
  const float* w_ln1  = (const float*)d_in[3];
  const float* w_q    = (const float*)d_in[4];
  const float* w_k    = (const float*)d_in[5];
  const float* w_v    = (const float*)d_in[6];
  const float* w_o    = (const float*)d_in[7];
  const float* w_ln2  = (const float*)d_in[8];
  const float* w_gu   = (const float*)d_in[9];
  const float* w_dn   = (const float*)d_in[10];
  float* out = (float*)d_out;

  char* ws = (char*)d_ws;
  size_t off = 0;
  auto alloc = [&](size_t bytes) -> void* {
    off = (off + 255) & ~(size_t)255;
    void* p = ws + off;
    off += bytes;
    return p;
  };

  const long n_wq  = (long)NH_  * HD_ * H_;   // 4M
  const long n_wk  = (long)KVH_ * HD_ * H_;   // 1M
  const long n_wv  = n_wk;
  const long n_wo  = (long)H_ * NH_ * HD_;    // 4M
  const long n_wgu = 2L * I_ * H_;            // 32M
  const long n_wdn = (long)H_ * I_;           // 16M

  u16*   wq_bf   = (u16*)alloc(n_wq  * 2);
  u16*   wk_bf   = (u16*)alloc(n_wk  * 2);
  u16*   wv_bf   = (u16*)alloc(n_wv  * 2);
  u16*   wo_bf   = (u16*)alloc(n_wo  * 2);
  u16*   wgu_bf  = (u16*)alloc(n_wgu * 2);
  u16*   wdn_bf  = (u16*)alloc(n_wdn * 2);
  u16*   ain_bf  = (u16*)alloc((size_t)TOK_ * H_ * 2);            // rmsnorm1 out
  float* q_f     = (float*)alloc((size_t)TOK_ * NH_  * HD_ * 4);
  float* k_f     = (float*)alloc((size_t)TOK_ * KVH_ * HD_ * 4);
  float* v_f     = (float*)alloc((size_t)TOK_ * KVH_ * HD_ * 4);
  u16*   qT_bf   = (u16*)alloc((size_t)TOK_ * NH_  * HD_ * 2);
  u16*   kT_bf   = (u16*)alloc((size_t)TOK_ * KVH_ * HD_ * 2);
  u16*   vT_bf   = (u16*)alloc((size_t)TOK_ * KVH_ * HD_ * 2);
  u16*   attn_bf = (u16*)alloc((size_t)TOK_ * NH_ * HD_ * 2);
  float* h1      = (float*)alloc((size_t)TOK_ * H_ * 4);          // h + attn_out
  u16*   mlp_bf  = (u16*)alloc((size_t)TOK_ * H_ * 2);
  float* gu_f    = (float*)alloc((size_t)TOK_ * 2 * I_ * 4);
  u16*   act_bf  = (u16*)alloc((size_t)TOK_ * I_ * 2);

  auto cvt = [&](const float* src, u16* dst, long n) {
    int blocks = (int)((n + 255) / 256);
    cvt_f32_to_bf16_kernel<<<blocks, 256, 0, stream>>>(src, dst, n);
  };
  cvt(w_q,  wq_bf,  n_wq);
  cvt(w_k,  wk_bf,  n_wk);
  cvt(w_v,  wv_bf,  n_wv);
  cvt(w_o,  wo_bf,  n_wo);
  cvt(w_gu, wgu_bf, n_wgu);
  cvt(w_dn, wdn_bf, n_wdn);

  // 1) rmsnorm -> bf16
  rmsnorm_bf16_kernel<<<TOK_, 256, 0, stream>>>(hidden, w_ln1, ain_bf);

  // 2) QKV projections
  gemm_bf16_kernel<<<dim3((NH_ * HD_) / 128, TOK_ / 128), 256, 0, stream>>>(
      ain_bf, wq_bf, q_f, nullptr, TOK_, NH_ * HD_, H_);
  gemm_bf16_kernel<<<dim3((KVH_ * HD_) / 128, TOK_ / 128), 256, 0, stream>>>(
      ain_bf, wk_bf, k_f, nullptr, TOK_, KVH_ * HD_, H_);
  gemm_bf16_kernel<<<dim3((KVH_ * HD_) / 128, TOK_ / 128), 256, 0, stream>>>(
      ain_bf, wv_bf, v_f, nullptr, TOK_, KVH_ * HD_, H_);

  // 3) RoPE + transpose to (B, heads, S, HD) bf16
  {
    long tq = (long)TOK_ * NH_ * (HD_ / 2);
    rope_transpose_kernel<<<(int)((tq + 255) / 256), 256, 0, stream>>>(
        q_f, sin_t, cos_t, qT_bf, NH_);
    long tk = (long)TOK_ * KVH_ * (HD_ / 2);
    rope_transpose_kernel<<<(int)((tk + 255) / 256), 256, 0, stream>>>(
        k_f, sin_t, cos_t, kT_bf, KVH_);
    long tv = (long)TOK_ * KVH_ * HD_;
    transpose_bf16_kernel<<<(int)((tv + 255) / 256), 256, 0, stream>>>(v_f, vT_bf);
  }

  // 4) causal flash attention -> (B,S,NH*HD) bf16
  flash_attn_kernel<<<dim3(S_ / 16, NH_, B_), 32, 0, stream>>>(qT_bf, kT_bf, vT_bf, attn_bf);

  // 5) O projection + residual -> h1
  gemm_bf16_kernel<<<dim3(H_ / 128, TOK_ / 128), 256, 0, stream>>>(
      attn_bf, wo_bf, h1, hidden, TOK_, H_, NH_ * HD_);

  // 6) rmsnorm2 -> bf16
  rmsnorm_bf16_kernel<<<TOK_, 256, 0, stream>>>(h1, w_ln2, mlp_bf);

  // 7) gate_up projection
  gemm_bf16_kernel<<<dim3((2 * I_) / 128, TOK_ / 128), 256, 0, stream>>>(
      mlp_bf, wgu_bf, gu_f, nullptr, TOK_, 2 * I_, H_);

  // 8) silu(gate)*up -> bf16
  {
    long ta = (long)TOK_ * I_;
    silu_mul_kernel<<<(int)((ta + 255) / 256), 256, 0, stream>>>(gu_f, act_bf);
  }

  // 9) down projection + residual -> out
  gemm_bf16_kernel<<<dim3(H_ / 128, TOK_ / 128), 256, 0, stream>>>(
      act_bf, wdn_bf, out, h1, TOK_, H_, I_);
}
